// CloudCrop_context_85787676770332
// MI455X (gfx1250) — matrix operand order repaired and checked by reading, not compiled, not run on hardware
//
#include <hip/hip_runtime.h>
#include <hip/hip_bf16.h>
#include <stdint.h>

#define BD 4
#define NP 1024
#define CD 512
#define NS 32
#define ROWS (BD*NP*NS)      /* 131072 */
#define K1   515
#define K1P  544             /* 17*32 */
#define K2P  512
#define OCH  512
#define RAD  0.05f
#define HMN  (-0.02f)
#define HMX  0.04f
#define EPSN 1e-5f

typedef __attribute__((ext_vector_type(16))) __bf16 v16bf;
typedef __attribute__((ext_vector_type(8)))  float  v8f;
typedef __attribute__((ext_vector_type(4)))  unsigned int uint4v;
typedef __attribute__((vector_size(16)))     int    a_v4i;   // matches builtin param type

#if defined(__has_builtin)
#  if __has_builtin(__builtin_amdgcn_global_load_async_to_lds_b128)
#    define HAVE_ASYNC_LDS 1
#  endif
#  if __has_builtin(__builtin_amdgcn_s_wait_asynccnt)
#    define HAVE_WAIT_ASYNC 1
#  endif
#endif

static __device__ __forceinline__ void tile_copy16(const unsigned short* __restrict__ g,
                                                   unsigned short* l) {
#if defined(HAVE_ASYNC_LDS)
  // direct memory->LDS, tracked by ASYNCcnt, no VGPR round trip
  __builtin_amdgcn_global_load_async_to_lds_b128(
      (__attribute__((address_space(1))) a_v4i*)g,
      (__attribute__((address_space(3))) a_v4i*)l, 0, 0);
#else
  *(uint4v*)l = *(const uint4v*)g;
#endif
}

static __device__ __forceinline__ void wait_async_fills() {
#if defined(HAVE_ASYNC_LDS)
#  if defined(HAVE_WAIT_ASYNC)
  __builtin_amdgcn_s_wait_asynccnt(0);
#  else
  asm volatile("s_wait_asynccnt 0x0" ::: "memory");
#  endif
#endif
}

static __device__ __forceinline__ unsigned short f2bf(float f) {
  union { float f; uint32_t u; } x; x.f = f;
  uint32_t u = x.u;
  uint32_t r = (u + 0x7FFFu + ((u >> 16) & 1u)) >> 16;   // RNE, ignores NaN edge
  return (unsigned short)r;
}

// ---------------------------------------------------------------- zero stats
__global__ void k_zero(float* p, int n) {
  for (int i = blockIdx.x * blockDim.x + threadIdx.x; i < n;
       i += gridDim.x * blockDim.x) p[i] = 0.0f;
}

// -------------------- weights fp32 -> bf16, [O][Kpad] row-major (= GEMM B [n][k])
__global__ void k_wconv(const float* __restrict__ W1, const float* __restrict__ W2,
                        unsigned short* __restrict__ W1t, unsigned short* __restrict__ W2t) {
  int i = blockIdx.x * blockDim.x + threadIdx.x;
  if (i < OCH * K1P) {
    int o = i / K1P, k = i % K1P;
    W1t[i] = (k < K1) ? f2bf(W1[o * K1 + k]) : (unsigned short)0;
  }
  if (i < OCH * K2P) {
    W2t[i] = f2bf(W2[i]);                 // already [O][512] row-major
  }
}

// -------------------------- cylinder query + gather + bf16 A1 build (1 wave/bm)
__global__ void k_build(const float* __restrict__ xyz, const float* __restrict__ feats,
                        const float* __restrict__ rot, unsigned short* __restrict__ A) {
  __shared__ int sIdx[NS];
  const int lane = threadIdx.x;
  const int bm = blockIdx.x;
  const int b = bm >> 10;
  const float cx = xyz[bm * 3 + 0], cy = xyz[bm * 3 + 1], cz = xyz[bm * 3 + 2];
  float R[9];
#pragma unroll
  for (int j = 0; j < 9; j++) R[j] = rot[(size_t)bm * 9 + j];

  int total = 0;
  for (int n0 = 0; n0 < NP && total < NS; n0 += 32) {
    int n = n0 + lane;
    const float* p = &xyz[((size_t)b * NP + n) * 3];
    float dx = p[0] - cx, dy = p[1] - cy, dz = p[2] - cz;
    float r0 = dx * R[0] + dy * R[3] + dz * R[6];
    float r1 = dx * R[1] + dy * R[4] + dz * R[7];
    float r2 = dx * R[2] + dy * R[5] + dz * R[8];
    bool pred = (r1 * r1 + r2 * r2 < RAD * RAD) && (r0 > HMN) && (r0 < HMX);
    unsigned int bal = __builtin_amdgcn_ballot_w32(pred);
    int rank = __popc(bal & ((1u << lane) - 1u));
    if (pred && (total + rank) < NS) sIdx[total + rank] = n;
    total += __popc(bal);
  }
  if (total > NS) total = NS;
  __syncthreads();
  if (total == 0) { if (lane == 0) sIdx[0] = bm & 1023; total = 1; }
  __syncthreads();
  int first = sIdx[0];
  if (lane >= total) sIdx[lane] = first;
  __syncthreads();

  const size_t rowbase0 = (size_t)bm * NS * K1P;
  for (int s = 0; s < NS; s++) {
    int n = sIdx[s];
    size_t rb = rowbase0 + (size_t)s * K1P;
    if (lane < 3) {
      const float* p = &xyz[((size_t)b * NP + n) * 3];
      float dx = p[0] - cx, dy = p[1] - cy, dz = p[2] - cz;
      float re = dx * R[lane] + dy * R[3 + lane] + dz * R[6 + lane];
      A[rb + lane] = f2bf(re * (1.0f / RAD));
    }
    const float* fb = &feats[(size_t)b * CD * NP + n];
    for (int c = lane; c < CD; c += 32)
      A[rb + 3 + c] = f2bf(fb[(size_t)c * NP]);
    if (lane < (K1P - 3 - CD)) A[rb + 3 + CD + lane] = 0;   // zero K padding
  }
}

// ------------------------------------------------------- WMMA bf16 tiled GEMM
// C[ROWS x 512] = A[ROWS x kpad] * Wt[512 x kpad]^T  (Wt row-major [n][k])
// block tile 128x128, 8 waves, wave tile 32x64 (2x4 WMMA 16x16 tiles)
// both tiles: double-buffered async memory->LDS; fused per-channel sum/sumsq.
__global__ void __launch_bounds__(256)
k_gemm(const unsigned short* __restrict__ A, const unsigned short* __restrict__ Wt,
       float* __restrict__ H, int kpad, float* __restrict__ s1, float* __restrict__ s2) {
  __shared__ __align__(16) unsigned short As[2][128 * 32];   // [row][k]
  __shared__ __align__(16) unsigned short Bs[2][128 * 32];   // [n][k]
  __shared__ float redS1[128], redS2[128];

  const int tid = threadIdx.x;
  const int lane = tid & 31, wave = tid >> 5;
  const int col0 = blockIdx.x * 128;
  const int row0 = blockIdx.y * 128;
  const int wrow = (wave >> 1) * 32;   // 0,32,64,96
  const int wcol = (wave & 1) * 64;    // 0,64

  if (tid < 128) { redS1[tid] = 0.0f; redS2[tid] = 0.0f; }

  v8f acc[2][4];
#pragma unroll
  for (int i = 0; i < 2; i++)
#pragma unroll
    for (int j = 0; j < 4; j++)
#pragma unroll
      for (int r = 0; r < 8; r++) acc[i][j][r] = 0.0f;

  const int ksteps = kpad >> 5;
  const int am = lane & 15, ag = lane >> 4;

  const unsigned short* aSrc = A  + (size_t)row0 * kpad;   // block's first A row
  const unsigned short* bSrc = Wt + (size_t)col0 * kpad;   // block's first W row

  // copy a 128-row x 32-half tile (row stride kpad) into LDS (row stride 32)
  auto fill = [&](const unsigned short* src, unsigned short* lds, int k0) {
#pragma unroll
    for (int i = 0; i < 2; i++) {
      int c = tid + 256 * i;              // 0..511 chunks of 16B
      int r = c >> 2, cc = (c & 3) * 8;   // 4 chunks per 32-half row
      tile_copy16(src + (size_t)r * kpad + k0 + cc, lds + r * 32 + cc);
    }
  };

  fill(aSrc, As[0], 0);
  fill(bSrc, Bs[0], 0);
  wait_async_fills();
  __syncthreads();

  for (int ks = 0; ks < ksteps; ks++) {
    const int cur = ks & 1;
    if (ks + 1 < ksteps) {                 // prefetch next tile into other buffer
      fill(aSrc, As[1 - cur], (ks + 1) * 32);
      fill(bSrc, Bs[1 - cur], (ks + 1) * 32);
    }

    union Frag { v16bf v; uint32_t u[8]; } af[2], bfr[4];
#pragma unroll
    for (int i = 0; i < 2; i++) {          // A frag: lanes 0-15 K=0..15, 16-31 K=16..31
      const uint32_t* rowp = (const uint32_t*)&As[cur][(wrow + i * 16 + am) * 32];
#pragma unroll
      for (int p = 0; p < 8; p++) {
        int k = ag * 16 + 2 * p;
        af[i].u[p] = rowp[k >> 1];
      }
    }
#pragma unroll
    for (int j = 0; j < 4; j++) {          // B frag: lanes 0-15 K=0..15, 16-31 K=16..31
      const uint32_t* colp = (const uint32_t*)&Bs[cur][(wcol + j * 16 + am) * 32];
#pragma unroll
      for (int p = 0; p < 8; p++)
        bfr[j].u[p] = colp[ag * 8 + p];
    }
#pragma unroll
    for (int i = 0; i < 2; i++)
#pragma unroll
      for (int j = 0; j < 4; j++)
        acc[i][j] = __builtin_amdgcn_wmma_f32_16x16x32_bf16(
            false, af[i].v, false, bfr[j].v, (short)0, acc[i][j], false, false);

    wait_async_fills();                    // next-tile async copies complete
    __syncthreads();                       // all waves' fills visible
  }

  // writeback + fused BN statistics (sum / sumsq per channel)
  const int cn = lane & 15, cg = lane >> 4;   // C/D layout per ISA
#pragma unroll
  for (int j = 0; j < 4; j++) {
    int col = wcol + j * 16 + cn;             // 0..127 within block
    float p1 = 0.0f, p2 = 0.0f;
#pragma unroll
    for (int i = 0; i < 2; i++)
#pragma unroll
      for (int r = 0; r < 8; r++) {
        float v = acc[i][j][r];
        int row = row0 + wrow + i * 16 + cg * 8 + r;
        H[(size_t)row * OCH + col0 + col] = v;
        p1 += v; p2 += v * v;
      }
    atomicAdd(&redS1[col], p1);               // ds_add_f32
    atomicAdd(&redS2[col], p2);
  }
  __syncthreads();
  if (tid < 128) {
    atomicAdd(&s1[col0 + tid], redS1[tid]);
    atomicAdd(&s2[col0 + tid], redS2[tid]);
  }
}

// --------------------------------------- BN + ReLU, repack to bf16 A2 [row][512]
__global__ void k_bnrelu(const float* __restrict__ H, const float* __restrict__ s1,
                         const float* __restrict__ s2, const float* __restrict__ g,
                         const float* __restrict__ bia, unsigned short* __restrict__ A2) {
  size_t i = (size_t)blockIdx.x * 256 + threadIdx.x;
  int c = (int)(i & 511);
  float mean = s1[c] * (1.0f / ROWS);
  float var  = s2[c] * (1.0f / ROWS) - mean * mean;
  float y = (H[i] - mean) * rsqrtf(var + EPSN) * g[c] + bia[c];
  A2[i] = f2bf(y > 0.0f ? y : 0.0f);
}

// ----------------------------------- BN + ReLU + max over S -> out (B,512,M)
__global__ void k_final(const float* __restrict__ H, const float* __restrict__ s1,
                        const float* __restrict__ s2, const float* __restrict__ g,
                        const float* __restrict__ bia, float* __restrict__ out) {
  const int bm = blockIdx.x;
  const int b = bm >> 10, m = bm & 1023;
  const size_t base = (size_t)bm * NS;
  for (int o = threadIdx.x; o < OCH; o += blockDim.x) {
    float mean = s1[o] * (1.0f / ROWS);
    float inv  = rsqrtf(s2[o] * (1.0f / ROWS) - mean * mean + EPSN);
    float gg = g[o], bb = bia[o];
    float mx = 0.0f;                       // relu outputs are >= 0
    for (int s = 0; s < NS; s++) {
      float y = (H[(base + s) * OCH + o] - mean) * inv * gg + bb;
      y = y > 0.0f ? y : 0.0f;
      mx = mx > y ? mx : y;
    }
    out[((size_t)b * OCH + o) * NP + m] = mx;
  }
}

extern "C" void kernel_launch(void* const* d_in, const int* in_sizes, int n_in,
                              void* d_out, int out_size, void* d_ws, size_t ws_size,
                              hipStream_t stream) {
  (void)in_sizes; (void)n_in; (void)out_size; (void)ws_size;
  const float* xyz   = (const float*)d_in[0];
  const float* feats = (const float*)d_in[1];
  const float* rot   = (const float*)d_in[2];
  const float* W1    = (const float*)d_in[3];
  const float* g1    = (const float*)d_in[4];
  const float* b1    = (const float*)d_in[5];
  const float* W2    = (const float*)d_in[6];
  const float* g2    = (const float*)d_in[7];
  const float* b2    = (const float*)d_in[8];
  float* out = (float*)d_out;

  char* ws = (char*)d_ws;
  unsigned short* Abuf = (unsigned short*)ws;                              // ROWS*K1P*2
  float*          Hbuf = (float*)(ws + (size_t)ROWS * K1P * 2);           // ROWS*512*4
  unsigned short* W1t  = (unsigned short*)((char*)Hbuf + (size_t)ROWS * OCH * 4);
  unsigned short* W2t  = (unsigned short*)((char*)W1t + (size_t)OCH * K1P * 2);
  float*          stat = (float*)((char*)W2t + (size_t)OCH * K2P * 2);    // 4*512 f32

  k_zero<<<8, 256, 0, stream>>>(stat, 4 * 512);
  k_wconv<<<(OCH * K1P + 255) / 256, 256, 0, stream>>>(W1, W2, W1t, W2t);
  k_build<<<BD * NP, 32, 0, stream>>>(xyz, feats, rot, Abuf);

  dim3 gg(OCH / 128, ROWS / 128);
  k_gemm<<<gg, 256, 0, stream>>>(Abuf, W1t, Hbuf, K1P, stat + 0, stat + 512);
  k_bnrelu<<<(unsigned)((size_t)ROWS * OCH / 256), 256, 0, stream>>>(
      Hbuf, stat + 0, stat + 512, g1, b1, Abuf);   // A2 reuses A region (stride 512)
  k_gemm<<<gg, 256, 0, stream>>>(Abuf, W2t, Hbuf, K2P, stat + 1024, stat + 1536);
  k_final<<<BD * NP, 256, 0, stream>>>(Hbuf, stat + 1024, stat + 1536, g2, b2, out);
}
